// QuantLinear_44805098832389
// MI455X (gfx1250) — matrix operand order repaired and checked by reading, not compiled, not run on hardware
//
#include <hip/hip_runtime.h>

// GPTQ 4-bit dequant + GEMM for MI455X (gfx1250, wave32, WMMA).
//   out[64,14336] = x[64,4096] @ (scales[g]*(w - z[g])) + bias
//
// Roofline: qweight (29.4 MB) dominates HBM traffic -> ~1.3 us floor at
// 23.3 TB/s; 7.5 GFLOP of f16 WMMA is the same order, so: stream qweight
// once, dequant in-register (0x6400 magic-bias -> v_pk_fma_f16), feed
// v_wmma_f32_16x16x32_f16.
//
// v2: x is pre-converted to f16 in d_ws by a prologue kernel (read once as
// f32, 512 KB f16 lives in the 192 MB L2). The GEMM kernel double-buffers
// the 64x128 x-tile in LDS using GLOBAL_LOAD_ASYNC_TO_LDS_B128 (ASYNCcnt),
// overlapping the tile DMA for group g+1 with dequant+WMMA of group g.
// The async builtin takes typed int4 pointers: (v4i AS1*, v4i AS3*, imm, imm).

#define IN_DIM   4096
#define OUT_DIM  14336
#define M_DIM    64
#define GROUP    128
#define NGROUPS  (IN_DIM / GROUP)
#define NTILE    64            // output columns per workgroup
#define THREADS  128           // 4 waves (wave32)
#define LDS_STRIDE 136         // 128 halves + 8 pad -> 272B row stride (bank-safe)

typedef __attribute__((ext_vector_type(16))) _Float16 v16h;
typedef __attribute__((ext_vector_type(8)))  _Float16 v8h;
typedef __attribute__((ext_vector_type(4)))  _Float16 v4h;
typedef __attribute__((ext_vector_type(2)))  _Float16 v2h;
typedef __attribute__((ext_vector_type(8)))  float    v8f;
typedef __attribute__((ext_vector_type(4)))  int      v4i;

typedef __attribute__((address_space(1))) v4i* gv4i_ptr;   // global int4*
typedef __attribute__((address_space(3))) v4i* lv4i_ptr;   // LDS int4*

#if __has_builtin(__builtin_amdgcn_global_load_async_to_lds_b128) && \
    __has_builtin(__builtin_amdgcn_s_wait_asynccnt)
#define USE_ASYNC 1
#else
#define USE_ASYNC 0
#endif

// ---------------- prologue: x f32 -> f16 into workspace ----------------
__global__ __launch_bounds__(256)
void convert_x_kernel(const float* __restrict__ x, _Float16* __restrict__ xh) {
  const int i = blockIdx.x * blockDim.x + threadIdx.x;  // one float4 each
  float4 f = reinterpret_cast<const float4*>(x)[i];
  v4h h = { (_Float16)f.x, (_Float16)f.y, (_Float16)f.z, (_Float16)f.w };
  reinterpret_cast<v4h*>(xh)[i] = h;
}

// Dequant 8 packed 4-bit values (one int32) into 8 f16 weights.
// f16 bits (0x6400 | v) == 1024+v exactly for v in [0,16), so
//   w = s*(v - z) = (1024+v)*s + (-(1024+z)*s)   -> one packed FMA per pair.
__device__ __forceinline__ v8h dequant8(unsigned q, v2h s2, v2h b2) {
  unsigned t0 = ( q         & 0x0000000Fu) | ((q << 12) & 0x000F0000u) | 0x64006400u;
  unsigned t1 = ((q >> 8)   & 0x0000000Fu) | ((q <<  4) & 0x000F0000u) | 0x64006400u;
  unsigned t2 = ((q >> 16)  & 0x0000000Fu) | ((q >>  4) & 0x000F0000u) | 0x64006400u;
  unsigned t3 = ((q >> 24)  & 0x0000000Fu) | ((q >> 12) & 0x000F0000u) | 0x64006400u;
  v2h p0 = __builtin_bit_cast(v2h, t0) * s2 + b2;
  v2h p1 = __builtin_bit_cast(v2h, t1) * s2 + b2;
  v2h p2 = __builtin_bit_cast(v2h, t2) * s2 + b2;
  v2h p3 = __builtin_bit_cast(v2h, t3) * s2 + b2;
  v8h r = { p0.x, p0.y, p1.x, p1.y, p2.x, p2.y, p3.x, p3.y };
  return r;
}

// Issue the 64x128-half tile copy for K-chunk k0 into LDS buffer `dst`.
// 128 threads: 2 per row, 64 halves (128 B = 8 x b128) each.
__device__ __forceinline__ void stage_tile(const _Float16* __restrict__ xh,
                                           _Float16* dst, int k0, int tid) {
  const int row = tid >> 1;
  const int seg = (tid & 1) * 64;
  const _Float16* src = xh + row * IN_DIM + k0 + seg;
  _Float16* d = dst + row * LDS_STRIDE + seg;
  #pragma unroll
  for (int i = 0; i < 8; ++i) {
#if USE_ASYNC
    __builtin_amdgcn_global_load_async_to_lds_b128(
        (gv4i_ptr)(src + 8 * i),
        (lv4i_ptr)(d + 8 * i),
        0, 0);
#else
    *reinterpret_cast<v8h*>(d + 8 * i) =
        *reinterpret_cast<const v8h*>(src + 8 * i);
#endif
  }
}

__global__ __launch_bounds__(THREADS)
void gptq_wmma_kernel(const _Float16* __restrict__ xh,
                      const int*   __restrict__ qweight,
                      const int*   __restrict__ qzeros,
                      const float* __restrict__ scales,
                      const float* __restrict__ bias,
                      float* __restrict__ out) {
  __shared__ _Float16 xs[2][M_DIM * LDS_STRIDE];   // 2 x 17,408 B (double buffer)

  const int tid   = threadIdx.x;
  const int lane  = tid & 31;
  const int wave  = tid >> 5;
  const int l16   = lane & 15;
  const int khalf = lane >> 4;                   // 0: K{0-15}, 1: K{16-31}

  const int col = blockIdx.x * NTILE + wave * 16 + l16;  // this lane's output column

  v8f acc[4] = {{}, {}, {}, {}};                 // 4 M-tiles of 16 rows

  // prime buffer 0 with group 0
  stage_tile(xh, &xs[0][0], 0, tid);

  for (int g = 0; g < NGROUPS; ++g) {
    const int k0  = g * GROUP;
    const int buf = g & 1;

    // kick off the next tile's DMA into the other buffer, then wait for
    // this tile's copies (in-order completion: wait <= 8 drains group g).
    if (g + 1 < NGROUPS) {
      stage_tile(xh, &xs[buf ^ 1][0], k0 + GROUP, tid);
#if USE_ASYNC
      __builtin_amdgcn_s_wait_asynccnt(8);
#endif
    } else {
#if USE_ASYNC
      __builtin_amdgcn_s_wait_asynccnt(0);
#endif
    }
    __syncthreads();   // tile g visible to all waves

    // per-group dequant constants for this lane's column
    const float s  = scales[g * OUT_DIM + col];
    const int   zq = ((qzeros[g * (OUT_DIM / 8) + (col >> 3)] >> ((col & 7) * 4)) & 15) + 1;
    const _Float16 hs = (_Float16)s;
    const _Float16 hb = (_Float16)(-s * (float)(1024 + zq));
    const v2h s2 = { hs, hs };
    const v2h b2 = { hb, hb };

    // hint the next group's qweight rows toward cache while we compute
    if (g + 1 < NGROUPS) {
      __builtin_prefetch(&qweight[(((k0 + GROUP) >> 3)) * OUT_DIM + col], 0, 1);
    }

    #pragma unroll
    for (int ks = 0; ks < GROUP / 32; ++ks) {
      const int kk = ks * 32;

      // ---- B fragment: 16 columns, K = k0+kk..+31 (16 K values per lane).
      const int row0 = (k0 + kk + khalf * 16) >> 3;
      const unsigned q0 = (unsigned)qweight[ row0      * OUT_DIM + col];
      const unsigned q1 = (unsigned)qweight[(row0 + 1) * OUT_DIM + col];
      v8h blo = dequant8(q0, s2, b2);
      v8h bhi = dequant8(q1, s2, b2);
      v16h bfrag = __builtin_shufflevector(blo, bhi,
          0, 1, 2, 3, 4, 5, 6, 7, 8, 9, 10, 11, 12, 13, 14, 15);

      // ---- 4 M-tiles share this B fragment
      #pragma unroll
      for (int mt = 0; mt < 4; ++mt) {
        const _Float16* ap = &xs[buf][(mt * 16 + l16) * LDS_STRIDE + kk + khalf * 16];
        v8h alo = *reinterpret_cast<const v8h*>(ap);        // K j..j+7
        v8h ahi = *reinterpret_cast<const v8h*>(ap + 8);    // K j+8..j+15
        v16h afrag = __builtin_shufflevector(alo, ahi,
            0, 1, 2, 3, 4, 5, 6, 7, 8, 9, 10, 11, 12, 13, 14, 15);
        acc[mt] = __builtin_amdgcn_wmma_f32_16x16x32_f16(
            false, afrag, false, bfrag, (short)0, acc[mt], false, false);
      }
    }

    __syncthreads();   // done reading xs[buf] before it is overwritten
  }

  // Epilogue: D layout -> lane 0-15: M = r, lane 16-31: M = 8 + r (N = col).
  const float bv = bias[col];
  const int rowadd = khalf * 8;
  #pragma unroll
  for (int mt = 0; mt < 4; ++mt) {
    #pragma unroll
    for (int r = 0; r < 8; ++r) {
      const int m = mt * 16 + rowadd + r;
      out[m * OUT_DIM + col] = acc[mt][r] + bv;
    }
  }
}

extern "C" void kernel_launch(void* const* d_in, const int* in_sizes, int n_in,
                              void* d_out, int out_size, void* d_ws, size_t ws_size,
                              hipStream_t stream) {
  const float* x    = (const float*)d_in[0];
  const int*   qw   = (const int*)  d_in[1];
  const int*   qz   = (const int*)  d_in[2];
  const float* sc   = (const float*)d_in[3];
  const float* bias = (const float*)d_in[4];
  // d_in[5] = g_idx; reference builds it as arange(IN)//GROUP, which this
  // kernel assumes directly (standard non-act-order layout).
  float* out = (float*)d_out;

  _Float16* xh = (_Float16*)d_ws;   // 64*4096 f16 = 512 KB scratch

  // prologue: convert x to f16 once (one float4 per thread)
  convert_x_kernel<<<dim3((M_DIM * IN_DIM / 4) / 256), dim3(256), 0, stream>>>(x, xh);

  // main GEMM: 224 workgroups x 4 waves
  gptq_wmma_kernel<<<dim3(OUT_DIM / NTILE), dim3(THREADS), 0, stream>>>(
      xh, qw, qz, sc, bias, out);
}
